// BitConv2dInferP2_17600775979458
// MI455X (gfx1250) — 8-bit matrix operand order repaired and checked by reading, compiled, not run on hardware
//
#include <hip/hip_runtime.h>
#include <stdint.h>


typedef int v8i __attribute__((ext_vector_type(8)));

#define PADW      114
#define ACT_BYTES (32ull * 114ull * 114ull * 64ull)

__global__ __launch_bounds__(256) void zero_ws(uint4* __restrict__ p, int n16) {
  int i = blockIdx.x * 256 + threadIdx.x;
  if (i < n16) p[i] = make_uint4(0u, 0u, 0u, 0u);
}

__global__ __launch_bounds__(256) void repack_w(const int* __restrict__ wq,
                                                int8_t* __restrict__ wpk) {
  int t = blockIdx.x * 256 + threadIdx.x;
  if (t >= 9 * 64 * 64) return;
  int cin  = t & 63;
  int cout = (t >> 6) & 63;
  int tap  = t >> 12;
  int kh = tap / 3, kw = tap % 3;
  int v = wq[((cout * 64 + cin) * 3 + kh) * 3 + kw];
  wpk[t] = (int8_t)v;
}

__global__ __launch_bounds__(256) void quantize_nchw_to_nhwc(
    const float* __restrict__ x, const int* __restrict__ act_exp,
    int8_t* __restrict__ act) {
  __shared__ int8_t tile[16 * 64];
  int blk = blockIdx.x;
  int wt = blk % 7;
  int h  = (blk / 7) % 112;
  int n  = blk / (7 * 112);
  int w0 = wt * 16;

  float inv_step = exp2f(-(float)act_exp[0]);

  int t  = threadIdx.x;
  int c  = t >> 2;
  int wq = t & 3;
  const float* xp =
      x + ((((size_t)n * 64 + c) * 112 + h) * 112 + w0 + wq * 4);
  float4 v = *(const float4*)xp;
  float f[4] = {v.x, v.y, v.z, v.w};
#pragma unroll
  for (int j = 0; j < 4; ++j) {
    float q = fminf(fmaxf(f[j], -1.0f), 1.0f);
    float r = rintf(q * inv_step);
    r = fminf(fmaxf(r, -127.0f), 127.0f);
    tile[(wq * 4 + j) * 64 + c] = (int8_t)(int)r;
  }
  __syncthreads();

  int pix = t >> 4;
  int c4  = (t & 15) * 4;
  int val = *(const int*)(tile + pix * 64 + c4);
  size_t dst =
      ((((size_t)n * PADW + (h + 1)) * PADW) + (w0 + pix + 1)) * 64 + c4;
  *(int*)(act + dst) = val;
}

__global__ __launch_bounds__(256) void conv_wmma(
    const int8_t* __restrict__ act, const int8_t* __restrict__ wpk,
    const int* __restrict__ s_exp, const float* __restrict__ bias,
    const int* __restrict__ act_exp, float* __restrict__ out) {
  __shared__ int8_t wlds[9 * 64 * 64];

  {
    const uint4* src = (const uint4*)wpk;
    uint4* dst = (uint4*)wlds;
    for (int i = threadIdx.x; i < 2304; i += 256) dst[i] = src[i];
  }
  __syncthreads();

  int lane = threadIdx.x & 31;
  int wave = threadIdx.x >> 5;
  int tile = blockIdx.x * 8 + wave;
  int wt = tile % 7;
  int h  = (tile / 7) % 112;
  int n  = tile / (7 * 112);
  int w0 = wt * 16;

  int nlo = lane & 15;
  int hi  = lane >> 4;

  int ae = act_exp[0];
  float scale[4], bv[4];
#pragma unroll
  for (int nt = 0; nt < 4; ++nt) {
    int cout = nt * 16 + nlo;
    scale[nt] = exp2f((float)(ae + s_exp[cout]));
    bv[nt]    = bias[cout];
  }

  v8i acc[4] = {};

#pragma unroll
  for (int kh = 0; kh < 3; ++kh) {
#pragma unroll
    for (int kw = 0; kw < 3; ++kw) {
      int tap = kh * 3 + kw;
      const int8_t* ap =
          act + ((((size_t)n * PADW + (h + kh)) * PADW) + (w0 + nlo + kw)) * 64 +
          hi * 16;
      int2 t0 = *(const int2*)(ap);
      int2 t1 = *(const int2*)(ap + 8);
      int2 t2 = *(const int2*)(ap + 32);
      int2 t3 = *(const int2*)(ap + 40);
      v8i a;
      a[0] = t0.x; a[1] = t0.y; a[2] = t1.x; a[3] = t1.y;
      a[4] = t2.x; a[5] = t2.y; a[6] = t3.x; a[7] = t3.y;

#pragma unroll
      for (int nt = 0; nt < 4; ++nt) {
        const int8_t* bp = wlds + tap * 4096 + (nt * 16 + nlo) * 64 + hi * 16;
        int4 b0 = *(const int4*)(bp);
        int4 b1 = *(const int4*)(bp + 32);
        v8i b;
        b[0] = b0.x; b[1] = b0.y; b[2] = b0.z; b[3] = b0.w;
        b[4] = b1.x; b[5] = b1.y; b[6] = b1.z; b[7] = b1.w;
        acc[nt] = __builtin_amdgcn_wmma_i32_16x16x64_iu8(
            true, a, true, b, acc[nt], false, false);
      }
    }
  }

#pragma unroll
  for (int nt = 0; nt < 4; ++nt) {
    int cout = nt * 16 + nlo;
    float f[8];
#pragma unroll
    for (int r = 0; r < 8; ++r)
      f[r] = (float)acc[nt][r] * scale[nt] + bv[nt];
    float* op =
        out + ((((size_t)n * 64 + cout) * 112 + h) * 112 + (w0 + hi * 8));
    *(float4*)(op)     = make_float4(f[0], f[1], f[2], f[3]);
    *(float4*)(op + 4) = make_float4(f[4], f[5], f[6], f[7]);
  }
}

extern "C" void kernel_launch(void* const* d_in, const int* in_sizes, int n_in,
                              void* d_out, int out_size, void* d_ws,
                              size_t ws_size, hipStream_t stream) {
  const float* x       = (const float*)d_in[0];
  const int*   w_q     = (const int*)d_in[1];
  const int*   s_exp   = (const int*)d_in[2];
  const float* bias    = (const float*)d_in[3];
  const int*   act_exp = (const int*)d_in[4];
  float*       out     = (float*)d_out;

  int8_t* act = (int8_t*)d_ws;
  int8_t* wpk = act + ACT_BYTES;

  int n16 = (int)(ACT_BYTES / 16);
  zero_ws<<<(n16 + 255) / 256, 256, 0, stream>>>((uint4*)act, n16);
  repack_w<<<(9 * 64 * 64 + 255) / 256, 256, 0, stream>>>(w_q, wpk);
  quantize_nchw_to_nhwc<<<32 * 112 * 7, 256, 0, stream>>>(x, act_exp, act);
  conv_wmma<<<(32 * 112 * 7) / 8, 256, 0, stream>>>(act, wpk, s_exp, bias,
                                                    act_exp, out);
}
